// MultiHeadAttention_25537875542146
// MI455X (gfx1250) — matrix operand order repaired and checked by reading, not compiled, not run on hardware
//
#include <hip/hip_runtime.h>
#include <hip/hip_bf16.h>

typedef unsigned short u16;
typedef unsigned int   u32;
typedef unsigned long long u64;
typedef __attribute__((ext_vector_type(16))) __bf16 v16bf;
typedef __attribute__((ext_vector_type(8)))  float  v8f;
typedef __attribute__((ext_vector_type(4)))  u32    u32x4;
typedef __attribute__((ext_vector_type(8)))  u32    u32x8;
typedef __attribute__((ext_vector_type(2)))  u32    u32x2;
typedef __attribute__((ext_vector_type(4)))  float  f32x4;

#define B_   2
#define S_   2048
#define H_   16
#define D_   1024
#define DK_  64
#define MTOT 4096   // B_*S_

// D# group1 word0: data_size=8B (3<<16); A adds pad_enable(1<<20),
// pad_interval=16dw (3<<22), pad_amount=4dw (3<<25): 64B row -> 80B LDS pitch
#define G1W0_A 0x06D30000u
#define G1W0_B 0x00030000u

// ---------- helpers ----------
__device__ __forceinline__ u16 f2bf(float f) {
  union { float f; u32 u; } x; x.f = f;
  u32 u = x.u + 0x7FFFu + ((x.u >> 16) & 1u);   // round-to-nearest-even
  return (u16)(u >> 16);
}
__device__ __forceinline__ u32x2 pack4_bf16(f32x4 d) {
  u32 a = (u32)f2bf(d.x) | ((u32)f2bf(d.y) << 16);
  u32 b = (u32)f2bf(d.z) | ((u32)f2bf(d.w) << 16);
  u32x2 r; r.x = a; r.y = b; return r;
}
// A fragment: frag[t]=A[m][half*16+t]
__device__ __forceinline__ v16bf load_frag_a(const u16* As, int lane, int mtile) {
  int m = mtile * 16 + (lane & 15);
  const u16* p = As + m * 40 + ((lane >> 4) << 4);
  union { v16bf v; u32x4 q[2]; } u;
  u.q[0] = *(const u32x4*)p;
  u.q[1] = *(const u32x4*)(p + 8);
  return u.v;
}
// B fragment: frag[t]=B[(lane/16)*16+t][n], contiguous in swizzled LDS
__device__ __forceinline__ v16bf load_frag_b(const u16* Bs, int lane, int ntile) {
  const u16* p = Bs + (ntile << 9) + (lane << 4);
  union { v16bf v; u32x4 q[2]; } u;
  u.q[0] = *(const u32x4*)p;
  u.q[1] = *(const u32x4*)(p + 8);
  return u.v;
}
__device__ __forceinline__ v8f wmma_bf16(v16bf a, v16bf b, v8f c) {
  return __builtin_amdgcn_wmma_f32_16x16x32_bf16(false, a, false, b, (short)0, c, false, false);
}
__device__ __forceinline__ u32 lds_off_of(const void* p) {
  return (u32)(__SIZE_TYPE__)p;   // low 32 bits of generic addr = LDS offset
}
// Tensor Data Mover: 2-D tile DMA global->LDS (D# groups 0/1; VADDR2/3 = NULL)
// dims/tiles in data_size(8B) units; stride0 = row pitch in 8B units
__device__ __forceinline__ void tdm_load_2d(u32 lds_addr, const u16* gptr, u32 g1w0,
                                            u32 dim0, u32 dim1, u32 tile0, u32 tile1,
                                            u32 stride0)
{
  u64 ga = (u64)(__SIZE_TYPE__)gptr;
  u32x4 g0;
  g0.x = 1u;                                 // count=1, user descriptor
  g0.y = lds_addr;                           // bits 63:32 lds_addr
  g0.z = (u32)ga;                            // global_addr[31:0]
  g0.w = (u32)(ga >> 32) | 0x80000000u;      // global_addr[56:32] | type=2
  u32x8 g1;
  g1[0] = g1w0;                              // mask=0, data_size, pad cfg
  g1[1] = (dim0 & 0xFFFFu) << 16;            // tensor_dim0[15:0]
  g1[2] = (dim0 >> 16) | ((dim1 & 0xFFFFu) << 16);
  g1[3] = (dim1 >> 16) | (tile0 << 16);      // tile_dim0
  g1[4] = tile1;                             // tile_dim1, tile_dim2=0
  g1[5] = stride0;                           // tensor_dim0_stride[31:0]
  g1[6] = 0u;
  g1[7] = 0u;
  asm volatile("tensor_load_to_lds %0, %1" :: "s"(g0), "s"(g1) : "memory");
}
template<int N>
__device__ __forceinline__ void tdm_wait() {
  __builtin_amdgcn_s_wait_tensorcnt((short)N);   // SIMM16 must be literal
}

// ---------- P0: cast q/k/v fp32 -> bf16 (natural layout) ----------
__global__ __launch_bounds__(256) void mha_castX(
    const float* __restrict__ q, const float* __restrict__ k, const float* __restrict__ v,
    u16* __restrict__ qb, u16* __restrict__ kb, u16* __restrict__ vb)
{
  const int z = blockIdx.z;
  const float* src = z == 0 ? q : z == 1 ? k : v;
  u16* dst = z == 0 ? qb : z == 1 ? kb : vb;
  size_t i = ((size_t)blockIdx.x * 256 + threadIdx.x) * 8;
  f32x4 a = *(const f32x4*)(src + i);
  f32x4 b = *(const f32x4*)(src + i + 4);
  union { u32x4 q4; u32x2 h[2]; } u;
  u.h[0] = pack4_bf16(a); u.h[1] = pack4_bf16(b);
  *(u32x4*)(dst + i) = u.q4;
}

// ---------- P1: cast weights fp32 -> bf16, WMMA-B swizzled global layout ----------
// element (k,n) -> ((n/16)*64 + k/16)*256 + (n%16)*16 + (k%16)
__global__ __launch_bounds__(256) void mha_castW(
    const float* __restrict__ Wq, const float* __restrict__ Wk,
    const float* __restrict__ Wv, const float* __restrict__ Wp,
    u16* __restrict__ Wqs, u16* __restrict__ Wks,
    u16* __restrict__ Wvs, u16* __restrict__ Wps)
{
  const int z = blockIdx.z;
  const float* W = z == 0 ? Wq : z == 1 ? Wk : z == 2 ? Wv : Wp;
  u16* Ws = z == 0 ? Wqs : z == 1 ? Wks : z == 2 ? Wvs : Wps;
  size_t i = ((size_t)blockIdx.x * 256 + threadIdx.x) * 8;
  int kk = (int)(i >> 10), n0 = (int)(i & 1023);
  f32x4 a = *(const f32x4*)(W + i);
  f32x4 b = *(const f32x4*)(W + i + 4);
  float vals[8] = {a.x, a.y, a.z, a.w, b.x, b.y, b.z, b.w};
#pragma unroll
  for (int t = 0; t < 8; t++) {
    int n = n0 + t;
    Ws[((size_t)(n >> 4) * 64 + (kk >> 4)) * 256 + (n & 15) * 16 + (kk & 15)] = f2bf(vals[t]);
  }
}

// ---------- K1: fused QKV projection (TDM-fed, double buffered) ----------
__global__ __launch_bounds__(256) void mha_qkv(
    const u16* __restrict__ qb, const u16* __restrict__ kb, const u16* __restrict__ vb,
    const u16* __restrict__ Wqs, const u16* __restrict__ Wks, const u16* __restrict__ Wvs,
    const float* __restrict__ bq, const float* __restrict__ bk, const float* __restrict__ bv,
    u16* __restrict__ Qh, u16* __restrict__ Khs, u16* __restrict__ Vhs)
{
  const int which = blockIdx.z;
  const u16* X    = which == 0 ? qb  : which == 1 ? kb  : vb;
  const u16* Wsw  = which == 0 ? Wqs : which == 1 ? Wks : Wvs;
  const float* bias = which == 0 ? bq : which == 1 ? bk : bv;

  const int row_base = blockIdx.x * 128;
  const int col_base = blockIdx.y * 128;
  const int tid = threadIdx.x, lane = tid & 31, wave = tid >> 5;
  const int wm = wave >> 2, wn = wave & 3;

  __shared__ __align__(16) u16 As[2][128 * 40];
  __shared__ __align__(16) u16 Bs[2][8 * 512];

  v8f acc[4][2];
#pragma unroll
  for (int i = 0; i < 4; i++)
#pragma unroll
    for (int j = 0; j < 2; j++) acc[i][j] = (v8f)0.0f;

  auto issue = [&](int kk, int buf) {
    tdm_load_2d(lds_off_of(As[buf]), X + (size_t)row_base * D_ + kk,
                G1W0_A, 8, 128, 8, 128, D_ * 2 / 8);
    tdm_load_2d(lds_off_of(Bs[buf]),
                Wsw + ((size_t)(col_base >> 4) * 64 + (kk >> 4)) * 256,
                G1W0_B, 128, 8, 128, 8, 4096);
  };
  const int T = D_ / 32;
  if (wave == 0) issue(0, 0);
  for (int t = 0; t < T; t++) {
    __syncthreads();                       // prior readers of buf[(t+1)&1] done
    if (wave == 0) {
      if (t + 1 < T) { issue((t + 1) * 32, (t + 1) & 1); tdm_wait<2>(); }
      else tdm_wait<0>();
    }
    __syncthreads();                       // slab t resident for all waves
    const u16* Asb = As[t & 1]; const u16* Bsb = Bs[t & 1];
    v16bf af[4], bf[2];
#pragma unroll
    for (int i = 0; i < 4; i++) af[i] = load_frag_a(Asb, lane, wm * 4 + i);
#pragma unroll
    for (int j = 0; j < 2; j++) bf[j] = load_frag_b(Bsb, lane, wn * 2 + j);
#pragma unroll
    for (int i = 0; i < 4; i++)
#pragma unroll
      for (int j = 0; j < 2; j++) acc[i][j] = wmma_bf16(af[i], bf[j], acc[i][j]);
  }

  const int half = lane >> 4, nl = lane & 15;
#pragma unroll
  for (int i = 0; i < 4; i++)
#pragma unroll
    for (int j = 0; j < 2; j++) {
      int col = col_base + (wn * 2 + j) * 16 + nl;
      float bv_ = bias[col];
      int h = col >> 6, dk = col & 63;
#pragma unroll
      for (int r = 0; r < 8; r++) {
        int row = row_base + (wm * 4 + i) * 16 + half * 8 + r;
        int bb = row >> 11, ss = row & (S_ - 1);
        u16 val = f2bf(acc[i][j][r] + bv_);
        size_t bh_base = (((size_t)bb * H_ + h)) * S_ * DK_;
        if (which == 0) {
          Qh[bh_base + (size_t)ss * DK_ + dk] = val;
        } else if (which == 1) {   // scores-B layout: blk=(ss/16)*4+(dk/16)
          Khs[bh_base + ((size_t)(ss >> 4) * 4 + (dk >> 4)) * 256 + (ss & 15) * 16 + (dk & 15)] = val;
        } else {                   // av-B layout: blk=(dk/16)*128+(ss/16)
          Vhs[bh_base + ((size_t)(dk >> 4) * 128 + (ss >> 4)) * 256 + (dk & 15) * 16 + (ss & 15)] = val;
        }
      }
    }
}

// ---------- K2: scores = Q K^T / 8, mask -> fp32 attn (pre-softmax) ----------
__global__ __launch_bounds__(256) void mha_scores(
    const u16* __restrict__ Qh, const u16* __restrict__ Khs,
    const int* __restrict__ mask, float* __restrict__ attn)
{
  const int bh = blockIdx.z, b = bh >> 4;
  const u16* Qb = Qh  + (size_t)bh * S_ * DK_;
  const u16* Kb = Khs + (size_t)bh * S_ * DK_;
  const int sq_base = blockIdx.x * 128, sk_base = blockIdx.y * 128;
  const int tid = threadIdx.x, lane = tid & 31, wave = tid >> 5;
  const int wm = wave >> 2, wn = wave & 3;

  __shared__ __align__(16) u16 As[2][128 * 40];
  __shared__ __align__(16) u16 Bs[2][8 * 512];

  v8f acc[4][2];
#pragma unroll
  for (int i = 0; i < 4; i++)
#pragma unroll
    for (int j = 0; j < 2; j++) acc[i][j] = (v8f)0.0f;

  auto issue = [&](int kk, int buf) {
    tdm_load_2d(lds_off_of(As[buf]), Qb + (size_t)sq_base * DK_ + kk,
                G1W0_A, 8, 128, 8, 128, DK_ * 2 / 8);
    tdm_load_2d(lds_off_of(Bs[buf]),
                Kb + ((size_t)(sk_base >> 4) * 4 + (kk >> 4)) * 256,
                G1W0_B, 128, 8, 128, 8, 256);
  };
  const int T = DK_ / 32;   // 2
  if (wave == 0) issue(0, 0);
#pragma unroll
  for (int t = 0; t < T; t++) {
    __syncthreads();
    if (wave == 0) {
      if (t + 1 < T) { issue((t + 1) * 32, (t + 1) & 1); tdm_wait<2>(); }
      else tdm_wait<0>();
    }
    __syncthreads();
    const u16* Asb = As[t & 1]; const u16* Bsb = Bs[t & 1];
    v16bf af[4], bf[2];
#pragma unroll
    for (int i = 0; i < 4; i++) af[i] = load_frag_a(Asb, lane, wm * 4 + i);
#pragma unroll
    for (int j = 0; j < 2; j++) bf[j] = load_frag_b(Bsb, lane, wn * 2 + j);
#pragma unroll
    for (int i = 0; i < 4; i++)
#pragma unroll
      for (int j = 0; j < 2; j++) acc[i][j] = wmma_bf16(af[i], bf[j], acc[i][j]);
  }

  const int half = lane >> 4, nl = lane & 15;
#pragma unroll
  for (int j = 0; j < 2; j++) {
    int sk = sk_base + (wn * 2 + j) * 16 + nl;
    int mv = mask[b * S_ + sk];
#pragma unroll
    for (int i = 0; i < 4; i++)
#pragma unroll
      for (int r = 0; r < 8; r++) {
        int sq = sq_base + (wm * 4 + i) * 16 + half * 8 + r;
        float val = mv ? acc[i][j][r] * 0.125f : -1.0e9f;
        attn[((size_t)bh * S_ + sq) * S_ + sk] = val;
      }
  }
}

// ---------- K3: row softmax over 2048, in place on d_out ----------
__global__ __launch_bounds__(256) void mha_softmax(float* __restrict__ attn)
{
  float* row = attn + (size_t)blockIdx.x * S_;
  const int t = threadIdx.x;
  __shared__ float sred[256];
  float vals[8]; float lmax = -3.4e38f;
#pragma unroll
  for (int i = 0; i < 8; i++) { vals[i] = row[t + (i << 8)]; lmax = fmaxf(lmax, vals[i]); }
  sred[t] = lmax; __syncthreads();
  for (int s = 128; s > 0; s >>= 1) { if (t < s) sred[t] = fmaxf(sred[t], sred[t + s]); __syncthreads(); }
  float rmax = sred[0]; __syncthreads();
  float lsum = 0.f;
#pragma unroll
  for (int i = 0; i < 8; i++) { vals[i] = __expf(vals[i] - rmax); lsum += vals[i]; }
  sred[t] = lsum; __syncthreads();
  for (int s = 128; s > 0; s >>= 1) { if (t < s) sred[t] += sred[t + s]; __syncthreads(); }
  float inv = 1.0f / sred[0];
#pragma unroll
  for (int i = 0; i < 8; i++) row[t + (i << 8)] = vals[i] * inv;
}

// ---------- K4: ctx = attn @ V  (B via TDM double buffer, A fp32->bf16) ----------
__global__ __launch_bounds__(256) void mha_av(
    const float* __restrict__ attn, const u16* __restrict__ Vhs, u16* __restrict__ ctx)
{
  const int bh = blockIdx.z, b = bh >> 4, h = bh & 15;
  const float* Ab = attn + (size_t)bh * S_ * S_;
  const u16*   Vb = Vhs  + (size_t)bh * S_ * DK_;
  const int sq_base = blockIdx.x * 128;
  const int tid = threadIdx.x, lane = tid & 31, wave = tid >> 5;
  const int wm = wave >> 1, wn = wave & 1;

  __shared__ __align__(16) u16 As[128 * 40];
  __shared__ __align__(16) u16 Bs[2][4 * 512];

  v8f acc[2][2];
#pragma unroll
  for (int i = 0; i < 2; i++)
#pragma unroll
    for (int j = 0; j < 2; j++) acc[i][j] = (v8f)0.0f;

  auto issueB = [&](int kk, int buf) {
    tdm_load_2d(lds_off_of(Bs[buf]), Vb + (size_t)(kk >> 4) * 256,
                G1W0_B, 128, 4, 128, 4, 8192);
  };
  const int T = S_ / 32;
  if (wave == 0) issueB(0, 0);
  for (int t = 0; t < T; t++) {
    const int kk = t * 32;
    __syncthreads();                        // readers of As & Bs[(t+1)&1] done
#pragma unroll
    for (int c = 0; c < 4; c++) {           // A: attn fp32 -> bf16 (in-loop convert)
      int f = c * 256 + tid; int row = f >> 3, c4 = f & 7;
      f32x4 d = *(const f32x4*)(Ab + (size_t)(sq_base + row) * S_ + kk + c4 * 4);
      *(u32x2*)&As[row * 40 + c4 * 4] = pack4_bf16(d);
    }
    if (wave == 0) {
      if (t + 1 < T) { issueB((t + 1) * 32, (t + 1) & 1); tdm_wait<1>(); }
      else tdm_wait<0>();
    }
    if (kk + 32 < S_)
      __builtin_prefetch(Ab + (size_t)(sq_base + (tid >> 1)) * S_ + kk + 32, 0, 0);
    __syncthreads();
    const u16* Bsb = Bs[t & 1];
    v16bf af[2], bf[2];
#pragma unroll
    for (int i = 0; i < 2; i++) af[i] = load_frag_a(As, lane, wm * 2 + i);
#pragma unroll
    for (int j = 0; j < 2; j++) bf[j] = load_frag_b(Bsb, lane, wn * 2 + j);
#pragma unroll
    for (int i = 0; i < 2; i++)
#pragma unroll
      for (int j = 0; j < 2; j++) acc[i][j] = wmma_bf16(af[i], bf[j], acc[i][j]);
  }

  const int half = lane >> 4, nl = lane & 15;
#pragma unroll
  for (int i = 0; i < 2; i++)
#pragma unroll
    for (int j = 0; j < 2; j++) {
      int d = (wn * 2 + j) * 16 + nl;
#pragma unroll
      for (int r = 0; r < 8; r++) {
        int sq = sq_base + (wm * 2 + i) * 16 + half * 8 + r;
        ctx[((size_t)(b * S_ + sq)) * D_ + h * DK_ + d] = f2bf(acc[i][j][r]);
      }
    }
}

// ---------- K5: Y = ctx @ Wp + bp + residual(q) -> fp32 ----------
__global__ __launch_bounds__(256) void mha_proj(
    const u16* __restrict__ ctx, const u16* __restrict__ Wps, const float* __restrict__ bp,
    const float* __restrict__ qres, float* __restrict__ Xres)
{
  const int row_base = blockIdx.x * 128, col_base = blockIdx.y * 128;
  const int tid = threadIdx.x, lane = tid & 31, wave = tid >> 5;
  const int wm = wave >> 2, wn = wave & 3;

  __shared__ __align__(16) u16 As[2][128 * 40];
  __shared__ __align__(16) u16 Bs[2][8 * 512];

  v8f acc[4][2];
#pragma unroll
  for (int i = 0; i < 4; i++)
#pragma unroll
    for (int j = 0; j < 2; j++) acc[i][j] = (v8f)0.0f;

  auto issue = [&](int kk, int buf) {
    tdm_load_2d(lds_off_of(As[buf]), ctx + (size_t)row_base * D_ + kk,
                G1W0_A, 8, 128, 8, 128, D_ * 2 / 8);
    tdm_load_2d(lds_off_of(Bs[buf]),
                Wps + ((size_t)(col_base >> 4) * 64 + (kk >> 4)) * 256,
                G1W0_B, 128, 8, 128, 8, 4096);
  };
  const int T = D_ / 32;
  if (wave == 0) issue(0, 0);
  for (int t = 0; t < T; t++) {
    __syncthreads();
    if (wave == 0) {
      if (t + 1 < T) { issue((t + 1) * 32, (t + 1) & 1); tdm_wait<2>(); }
      else tdm_wait<0>();
    }
    __syncthreads();
    const u16* Asb = As[t & 1]; const u16* Bsb = Bs[t & 1];
    v16bf af[4], bf[2];
#pragma unroll
    for (int i = 0; i < 4; i++) af[i] = load_frag_a(Asb, lane, wm * 4 + i);
#pragma unroll
    for (int j = 0; j < 2; j++) bf[j] = load_frag_b(Bsb, lane, wn * 2 + j);
#pragma unroll
    for (int i = 0; i < 4; i++)
#pragma unroll
      for (int j = 0; j < 2; j++) acc[i][j] = wmma_bf16(af[i], bf[j], acc[i][j]);
  }

  const int half = lane >> 4, nl = lane & 15;
#pragma unroll
  for (int i = 0; i < 4; i++)
#pragma unroll
    for (int j = 0; j < 2; j++) {
      int col = col_base + (wn * 2 + j) * 16 + nl;
      float bv_ = bp[col];
#pragma unroll
      for (int r = 0; r < 8; r++) {
        int row = row_base + (wm * 4 + i) * 16 + half * 8 + r;
        size_t idx = (size_t)row * D_ + col;
        Xres[idx] = acc[i][j][r] + bv_ + qres[idx];
      }
    }
}

// ---------- K6: LayerNorm over 1024 ----------
__global__ __launch_bounds__(256) void mha_ln(
    const float* __restrict__ X, const float* __restrict__ gamma,
    const float* __restrict__ beta, float* __restrict__ out)
{
  const float* x = X + (size_t)blockIdx.x * D_;
  float* o = out + (size_t)blockIdx.x * D_;
  const int t = threadIdx.x;
  __shared__ float s1[256], s2[256];
  float v[4]; float sum = 0.f, sq = 0.f;
#pragma unroll
  for (int i = 0; i < 4; i++) { v[i] = x[t + (i << 8)]; sum += v[i]; sq += v[i] * v[i]; }
  s1[t] = sum; s2[t] = sq; __syncthreads();
  for (int s = 128; s > 0; s >>= 1) {
    if (t < s) { s1[t] += s1[t + s]; s2[t] += s2[t + s]; }
    __syncthreads();
  }
  float mu  = s1[0] * (1.0f / D_);
  float var = s2[0] * (1.0f / D_) - mu * mu;
  float rs  = rsqrtf(var + 1e-5f);
#pragma unroll
  for (int i = 0; i < 4; i++) {
    int c = t + (i << 8);
    o[c] = (v[i] - mu) * rs * gamma[c] + beta[c];
  }
}

// ---------- launch ----------
extern "C" void kernel_launch(void* const* d_in, const int* in_sizes, int n_in,
                              void* d_out, int out_size, void* d_ws, size_t ws_size,
                              hipStream_t stream) {
  (void)in_sizes; (void)n_in; (void)out_size; (void)ws_size;
  const float* q    = (const float*)d_in[0];
  const float* k    = (const float*)d_in[1];
  const float* v    = (const float*)d_in[2];
  const int*   mask = (const int*)  d_in[3];
  const float* Wq   = (const float*)d_in[4];
  const float* bq   = (const float*)d_in[5];
  const float* Wk   = (const float*)d_in[6];
  const float* bk   = (const float*)d_in[7];
  const float* Wv   = (const float*)d_in[8];
  const float* bv   = (const float*)d_in[9];
  const float* Wp   = (const float*)d_in[10];
  const float* bp   = (const float*)d_in[11];
  const float* gamma= (const float*)d_in[12];
  const float* beta = (const float*)d_in[13];

  float* out  = (float*)d_out;
  float* attn = out + (size_t)B_ * S_ * D_;      // tuple: (out, attn) flat

  char* w = (char*)d_ws;
  u16* qb   = (u16*)w;   w += (size_t)8 << 20;   // [B*S, D] bf16
  u16* kb   = (u16*)w;   w += (size_t)8 << 20;
  u16* vb   = (u16*)w;   w += (size_t)8 << 20;
  u16* Wqs  = (u16*)w;   w += (size_t)2 << 20;   // swizzled weights, bf16
  u16* Wks  = (u16*)w;   w += (size_t)2 << 20;
  u16* Wvs  = (u16*)w;   w += (size_t)2 << 20;
  u16* Wps  = (u16*)w;   w += (size_t)2 << 20;
  u16* Qh   = (u16*)w;   w += (size_t)8 << 20;   // [B,H,S,DK] natural
  u16* Khs  = (u16*)w;   w += (size_t)8 << 20;   // [B,H] swizzled scores-B
  u16* Vhs  = (u16*)w;   w += (size_t)8 << 20;   // [B,H] swizzled av-B
  u16* ctx  = (u16*)w;   w += (size_t)8 << 20;   // [B,S,D] bf16
  float* Xr = (float*)w;                          // [B,S,D] fp32 residual sum

  mha_castX  <<<dim3(MTOT*D_/(256*8), 1, 3), 256, 0, stream>>>(q,k,v,qb,kb,vb);
  mha_castW  <<<dim3(D_*D_/(256*8), 1, 4), 256, 0, stream>>>(Wq,Wk,Wv,Wp,Wqs,Wks,Wvs,Wps);
  mha_qkv    <<<dim3(MTOT/128, D_/128, 3), 256, 0, stream>>>(qb,kb,vb,Wqs,Wks,Wvs,bq,bk,bv,Qh,Khs,Vhs);
  mha_scores <<<dim3(S_/128, S_/128, B_*H_), 256, 0, stream>>>(Qh,Khs,mask,attn);
  mha_softmax<<<dim3(B_*H_*S_), 256, 0, stream>>>(attn);
  mha_av     <<<dim3(S_/128, 1, B_*H_), 256, 0, stream>>>(attn,Vhs,ctx);
  mha_proj   <<<dim3(MTOT/128, D_/128), 256, 0, stream>>>(ctx,Wps,bp,q,Xr);
  mha_ln     <<<dim3(MTOT), 256, 0, stream>>>(Xr,gamma,beta,out);
}
